// MiLSTM_12799002542193
// MI455X (gfx1250) — matrix operand order repaired and checked by reading, not compiled, not run on hardware
//
#include <hip/hip_runtime.h>
#include <hip/hip_bf16.h>

#define B_ 256
#define T_ 256
#define F_ 128
#define H_ 512

typedef __attribute__((ext_vector_type(16))) _Float16 v16h;
typedef __attribute__((ext_vector_type(8)))  _Float16 v8h;
typedef __attribute__((ext_vector_type(8)))  float    v8f;
typedef __attribute__((ext_vector_type(4)))  unsigned int v4u;
typedef __attribute__((ext_vector_type(8)))  int      v8i;
typedef __attribute__((ext_vector_type(4)))  int      v4i;

// Unified phase geometry: every staged tile is 10 gates x 16 rows x 128 f16,
// rows padded by 4 dwords (row stride 68 dwords -> +4 banks per row).
#define PH_ROW_BYTES   272                   // 128 f16 (64 dw) + 16B pad
#define PH_GATE_BYTES  (16 * PH_ROW_BYTES)   // 4352
#define PH_BUF_BYTES   (10 * PH_GATE_BYTES)  // 43520
#define LDS_BYTES      (2 * PH_BUF_BYTES)    // 87040 (double-buffered)

// ---------------------------------------------------------------------------
// Parameter packing (transpose to [n][k] f16, WMMA-B friendly), biases, alphas.
// Gate order: f, o, c, cp, cn, ci, i, ip, in, ii.
// Gate ci's recurrent weight is Wcph (bug preserved from reference).
// ---------------------------------------------------------------------------
struct PackArgs {
    const float* wh[10];
    const float* wy[10];
    const float* bias[10];
    const float* alpha[4];
    _Float16* WhT;      // [10][H][H]
    _Float16* WyT;      // [10][H][F]
    float* biasPack;    // [10][H]
    float* alphaPack;   // [4]
};

__global__ void milstm_pack_params(PackArgs a) {
    const int n   = blockIdx.x;
    const int g   = blockIdx.y;
    const int tid = threadIdx.x;
    for (int k = tid; k < H_; k += blockDim.x)
        a.WhT[((size_t)g * H_ + n) * H_ + k] = (_Float16)a.wh[g][(size_t)k * H_ + n];
    if (tid < F_)
        a.WyT[((size_t)g * H_ + n) * F_ + tid] = (_Float16)a.wy[g][(size_t)tid * H_ + n];
    if (tid == 0)
        a.biasPack[g * H_ + n] = a.bias[g][n];
    if (g == 0 && n == 0 && tid < 4)
        a.alphaPack[tid] = a.alpha[tid][0];
}

__global__ void milstm_convert_x(const float* __restrict__ y, const float* __restrict__ p,
                                 const float* __restrict__ nn, const float* __restrict__ ix,
                                 _Float16* __restrict__ o, int n) {
    int i = blockIdx.x * blockDim.x + threadIdx.x;
    int stride = gridDim.x * blockDim.x;
    for (; i < n; i += stride) {
        o[i]                 = (_Float16)y[i];
        o[(size_t)n + i]     = (_Float16)p[i];
        o[(size_t)2 * n + i] = (_Float16)nn[i];
        o[(size_t)3 * n + i] = (_Float16)ix[i];
    }
}

__global__ void milstm_init_state(_Float16* __restrict__ h, float* __restrict__ c,
                                  int nh, int nc) {
    int i = blockIdx.x * blockDim.x + threadIdx.x;
    if (i < nh) h[i] = (_Float16)0.0f;
    if (i < nc) c[i] = 0.0f;
}

// ---------------------------------------------------------------------------
// TDM: iterated 2D tile load Global->LDS; one instruction walks all 10 gate
// matrices (iterate_enable, address increments), padding LDS rows.
// D# layout per CDNA5 ISA.
// ---------------------------------------------------------------------------
__device__ __forceinline__ void tdm_load_iter(
    unsigned lds_off, const _Float16* gptr,
    unsigned tile_d0, unsigned tile_d1,
    unsigned tensor_d0, unsigned tensor_d1, unsigned row_stride,
    unsigned pad_interval_code, unsigned pad_amount_code,
    unsigned lds_inc_elems, unsigned gaddr_inc_elems, unsigned iters_m1)
{
    unsigned long long ga = (unsigned long long)(size_t)gptr;
    v4u g0;
    g0[0] = 1u;                                        // count=1 (valid), user mode
    g0[1] = lds_off;                                   // lds_addr (bytes)
    g0[2] = (unsigned)(ga & 0xffffffffu);              // global_addr lo
    g0[3] = (unsigned)((ga >> 32) & 0x01ffffffu) | (2u << 30);  // addr hi | type=2
    v8i g1;
    g1[0] = (int)((1u << 16)                           // data_size = 2B
                | (1u << 19)                           // iterate_enable
                | (1u << 20)                           // pad_enable
                | (pad_interval_code << 22)
                | (pad_amount_code << 25));
    g1[1] = (int)((tensor_d0 & 0xffffu) << 16);        // abar=0 | tensor_dim0 lo16
    g1[2] = (int)((tensor_d0 >> 16) | ((tensor_d1 & 0xffffu) << 16));
    g1[3] = (int)((tensor_d1 >> 16) | (tile_d0 << 16));
    g1[4] = (int)(tile_d1 & 0xffffu);                  // tile_dim1 | tile_dim2=0
    g1[5] = (int)row_stride;                           // tensor_dim0_stride lo32
    g1[6] = 0;                                         // stride0 hi | stride1 lo
    g1[7] = 0;
    v4i g2;
    g2[0] = 0;                                         // tensor_dim2
    g2[1] = (int)lds_inc_elems;                        // lds_addr_increment
    g2[2] = (int)gaddr_inc_elems;                      // global_addr_increment lo32
    g2[3] = (int)(iters_m1 << 16);                     // inc hi | iterate_count
    v4i g3 = {0, 0, 0, 0};
#if defined(__clang_major__) && (__clang_major__ >= 23)
    v8i g4 = {0, 0, 0, 0, 0, 0, 0, 0};
    __builtin_amdgcn_tensor_load_to_lds(g0, g1, g2, g3, g4, 0);
#else
    __builtin_amdgcn_tensor_load_to_lds(g0, g1, g2, g3, 0);
#endif
}

__device__ __forceinline__ float sig_(float x)  { return 1.0f / (1.0f + __expf(-x)); }
__device__ __forceinline__ float tanh_(float x) { return 1.0f - 2.0f / (__expf(2.0f * x) + 1.0f); }
__device__ __forceinline__ v16h cat8(v8h lo, v8h hi) {
    return __builtin_shufflevector(lo, hi, 0,1,2,3,4,5,6,7,8,9,10,11,12,13,14,15);
}

// One staged phase: 4 K-chunks of 32. For each chunk, stage ALL 10 gate B
// fragments from LDS first (20 ds_load_b128 issued back-to-back), then run
// the 10 WMMAs -- lets the scheduler retire dscnt waits incrementally.
__device__ __forceinline__ void compute_phase_h(
    v8f (&acc)[10], const unsigned char* buf, const _Float16* hseg,
    int l15, int half)
{
#pragma unroll
    for (int kcl = 0; kcl < 4; ++kcl) {
        const int ka = kcl * 32 + half * 8;
        v16h amat = cat8(*(const v8h*)(hseg + ka), *(const v8h*)(hseg + ka + 16));
        const unsigned fo = (unsigned)l15 * PH_ROW_BYTES
                          + (unsigned)(kcl * 32 + half * 8) * 2;
        v16h bm[10];
#pragma unroll
        for (int g = 0; g < 10; ++g) {
            const unsigned char* bp = buf + g * PH_GATE_BYTES + fo;
            bm[g] = cat8(*(const v8h*)bp, *(const v8h*)(bp + 32));
        }
#pragma unroll
        for (int g = 0; g < 10; ++g)
            acc[g] = __builtin_amdgcn_wmma_f32_16x16x32_f16(
                false, amat, false, bm[g], (short)0, acc[g], false, false);
    }
}

__device__ __forceinline__ void compute_phase_y(
    v8f (&acc)[10], const unsigned char* buf, const _Float16* x16, size_t xrow,
    int l15, int half)
{
    constexpr int srcmap[10] = {0,0,0,1,2,3,0,0,0,0};   // y,y,y,p,n,ix,y,y,y,y
#pragma unroll
    for (int kcl = 0; kcl < 4; ++kcl) {
        const int ka = kcl * 32 + half * 8;
        v16h ax[4];
#pragma unroll
        for (int s = 0; s < 4; ++s) {
            const _Float16* xp = x16 + (size_t)s * B_ * T_ * F_ + xrow;
            ax[s] = cat8(*(const v8h*)(xp + ka), *(const v8h*)(xp + ka + 16));
        }
        const unsigned fo = (unsigned)l15 * PH_ROW_BYTES
                          + (unsigned)(kcl * 32 + half * 8) * 2;
        v16h bm[10];
#pragma unroll
        for (int g = 0; g < 10; ++g) {
            const unsigned char* bp = buf + g * PH_GATE_BYTES + fo;
            bm[g] = cat8(*(const v8h*)bp, *(const v8h*)(bp + 32));
        }
#pragma unroll
        for (int g = 0; g < 10; ++g)
            acc[g] = __builtin_amdgcn_wmma_f32_16x16x32_f16(
                false, ax[srcmap[g]], false, bm[g], (short)0, acc[g], false, false);
    }
}

// ---------------------------------------------------------------------------
// Fused per-timestep kernel. Grid: (nt = 32, mh = 2); block: 8 waves, wave w
// owns batch tile mt = mh*8+w. Five TDM-staged phases (4x Wh K-quarters + Wy),
// LDS double-buffered so the next TDM is in flight while WMMAs consume the
// current buffer; wave 0 waits tensorcnt<=1, then tensorcnt 0 before Wy.
// ---------------------------------------------------------------------------
__global__ __launch_bounds__(256) void milstm_step(
    const _Float16* __restrict__ h_in,     // [B][H] f16 (h_{t-1})
    _Float16* __restrict__ h_out,          // [B][H] f16 (h_t)
    float* __restrict__ cbuf,              // [B][H] f32
    const _Float16* __restrict__ WhT,      // [10][H][H]
    const _Float16* __restrict__ WyT,      // [10][H][F]
    const _Float16* __restrict__ x16,      // [4][B][T][F]
    const float* __restrict__ biasPack,    // [10][H]
    const float* __restrict__ alphaPack,   // [4]
    float* __restrict__ out,               // hidden_seqs | h_T | c_T
    int t)
{
    __shared__ __attribute__((aligned(16))) unsigned char wlds[LDS_BYTES];

    const int tid  = threadIdx.x;
    const int lane = tid & 31;
    const int wave = tid >> 5;                 // 0..7
    const int l15  = lane & 15;
    const int half = lane >> 4;
    const int nt   = blockIdx.x;               // 0..31
    const int mt   = blockIdx.y * 8 + wave;    // 0..15
    const int nCol = nt * 16 + l15;
    const int aRow = mt * 16 + l15;
    const unsigned lds_base = (unsigned)(size_t)(void*)wlds;

    const v8f zero = {0.f,0.f,0.f,0.f,0.f,0.f,0.f,0.f};
    v8f acc[10];
#pragma unroll
    for (int g = 0; g < 10; ++g) acc[g] = zero;

    const _Float16* hrow = h_in + (size_t)aRow * H_;
    const size_t xrow = ((size_t)aRow * T_ + t) * F_;

    // TDM issue helpers (wave 0 only issues; geometry identical across phases)
    auto issue_wh = [&](int p, int buf) {
        tdm_load_iter(lds_base + (unsigned)buf * PH_BUF_BYTES,
                      WhT + (size_t)nt * 16 * H_ + (size_t)p * 128,
                      /*tile*/128, 16, /*tensor*/H_, 10 * H_, /*row_stride*/H_,
                      /*pad 64dw*/5, /*pad 4dw*/3,
                      PH_GATE_BYTES / 2, H_ * H_, 9);
    };
    auto issue_wy = [&](int buf) {
        tdm_load_iter(lds_base + (unsigned)buf * PH_BUF_BYTES,
                      WyT + (size_t)nt * 16 * F_,
                      128, 16, F_, 10 * H_, F_,
                      5, 3, PH_GATE_BYTES / 2, H_ * F_, 9);
    };

    // prologue:两 loads in flight; wait for the first only
    if (tid < 32) {
        issue_wh(0, 0);
        issue_wh(1, 1);
        __builtin_amdgcn_s_wait_tensorcnt(1);
    }
    __syncthreads();
    compute_phase_h(acc, wlds + 0 * PH_BUF_BYTES, hrow + 0 * 128, l15, half);
    __syncthreads();                               // buf0 free
    if (tid < 32) { issue_wh(2, 0); __builtin_amdgcn_s_wait_tensorcnt(1); }
    __syncthreads();                               // L1 complete
    compute_phase_h(acc, wlds + 1 * PH_BUF_BYTES, hrow + 1 * 128, l15, half);
    __syncthreads();                               // buf1 free
    if (tid < 32) { issue_wh(3, 1); __builtin_amdgcn_s_wait_tensorcnt(1); }
    __syncthreads();                               // L2 complete
    compute_phase_h(acc, wlds + 0 * PH_BUF_BYTES, hrow + 2 * 128, l15, half);
    __syncthreads();                               // buf0 free
    if (tid < 32) { issue_wy(0); __builtin_amdgcn_s_wait_tensorcnt(1); }
    __syncthreads();                               // L3 complete
    compute_phase_h(acc, wlds + 1 * PH_BUF_BYTES, hrow + 3 * 128, l15, half);
    __syncthreads();
    if (tid < 32) { __builtin_amdgcn_s_wait_tensorcnt(0); }
    __syncthreads();                               // Wy complete
    compute_phase_y(acc, wlds + 0 * PH_BUF_BYTES, x16, xrow, l15, half);

    // ---- elementwise gate math + state update ----
    float bg[10];
#pragma unroll
    for (int g = 0; g < 10; ++g) bg[g] = biasPack[g * H_ + nCol];
    const float a_t  = alphaPack[0];
    const float a_pt = alphaPack[1];
    const float a_nt = alphaPack[2];
    const float a_it = alphaPack[3];

#pragma unroll
    for (int r = 0; r < 8; ++r) {
        const int m = mt * 16 + r + half * 8;   // C/D layout: VGPR r, lane half
        const float f_t = sig_ (acc[0][r] + bg[0]);
        const float o_t = sig_ (acc[1][r] + bg[1]);
        const float c_t = tanh_(acc[2][r] + bg[2]);
        const float c_p = tanh_(acc[3][r] + bg[3]);
        const float c_n = tanh_(acc[4][r] + bg[4]);
        const float c_i = tanh_(acc[5][r] + bg[5]);
        const float i_t = sig_ (acc[6][r] + bg[6]);
        const float i_p = sig_ (acc[7][r] + bg[7]);
        const float i_n = sig_ (acc[8][r] + bg[8]);
        const float i_i = sig_ (acc[9][r] + bg[9]);
        const float L = a_t * c_t * i_t + a_pt * c_p * i_p
                      + a_nt * c_n * i_n + a_it * c_i * i_i;
        const int sidx = m * H_ + nCol;
        const float c_old = cbuf[sidx];
        const float c_new = c_old * f_t + L;
        const float h_new = tanh_(c_new) * o_t;
        cbuf[sidx]  = c_new;
        h_out[sidx] = (_Float16)h_new;
        // streaming output: keep it out of L2 (weights stay resident)
        __builtin_nontemporal_store(h_new, &out[((size_t)m * T_ + t) * H_ + nCol]);
        if (t == T_ - 1) {
            out[(size_t)B_ * T_ * H_ + sidx] = h_new;                        // h_T
            out[(size_t)B_ * T_ * H_ + (size_t)B_ * H_ + sidx] = c_new;      // c_T
        }
    }
}

// ---------------------------------------------------------------------------
extern "C" void kernel_launch(void* const* d_in, const int* in_sizes, int n_in,
                              void* d_out, int out_size, void* d_ws, size_t ws_size,
                              hipStream_t stream) {
    (void)in_sizes; (void)n_in; (void)out_size; (void)ws_size;
    const float* y  = (const float*)d_in[0];
    const float* pp = (const float*)d_in[1];
    const float* nn = (const float*)d_in[2];
    const float* ix = (const float*)d_in[3];

    static const int WH_IDX[10] = {4, 7, 10, 13, 16, /*ci uses Wcph:*/13, 22, 25, 28, 31};
    static const int WY_IDX[10] = {5, 8, 11, 14, 17, 20, 23, 26, 29, 32};
    static const int BI_IDX[10] = {6, 9, 12, 15, 18, 21, 24, 27, 30, 33};
    static const int AL_IDX[4]  = {34, 35, 36, 37};

    char* base = (char*)d_ws;
    size_t off = 0;
    auto carve = [&](size_t bytes) -> void* {
        void* p = base + off;
        off = (off + bytes + 255) & ~(size_t)255;
        return p;
    };
    _Float16* WhT      = (_Float16*)carve((size_t)10 * H_ * H_ * sizeof(_Float16));
    _Float16* WyT      = (_Float16*)carve((size_t)10 * H_ * F_ * sizeof(_Float16));
    float*    biasPack = (float*)   carve((size_t)10 * H_ * sizeof(float));
    float*    alphaPk  = (float*)   carve(4 * sizeof(float));
    _Float16* x16      = (_Float16*)carve((size_t)4 * B_ * T_ * F_ * sizeof(_Float16));
    _Float16* h16      = (_Float16*)carve((size_t)2 * B_ * H_ * sizeof(_Float16));
    float*    cbuf     = (float*)   carve((size_t)B_ * H_ * sizeof(float));

    PackArgs pa;
    for (int g = 0; g < 10; ++g) {
        pa.wh[g]   = (const float*)d_in[WH_IDX[g]];
        pa.wy[g]   = (const float*)d_in[WY_IDX[g]];
        pa.bias[g] = (const float*)d_in[BI_IDX[g]];
    }
    for (int i = 0; i < 4; ++i) pa.alpha[i] = (const float*)d_in[AL_IDX[i]];
    pa.WhT = WhT; pa.WyT = WyT; pa.biasPack = biasPack; pa.alphaPack = alphaPk;

    milstm_pack_params<<<dim3(H_, 10), 256, 0, stream>>>(pa);

    const int nX = B_ * T_ * F_;
    milstm_convert_x<<<dim3((nX + 255) / 256), 256, 0, stream>>>(y, pp, nn, ix, x16, nX);

    const int nh = 2 * B_ * H_, nc = B_ * H_;
    milstm_init_state<<<dim3((nh + 255) / 256), 256, 0, stream>>>(h16, cbuf, nh, nc);

    float* out = (float*)d_out;
    for (int t = 0; t < T_; ++t) {
        const _Float16* hin  = h16 + (size_t)(t & 1) * B_ * H_;
        _Float16*       hout = h16 + (size_t)((t + 1) & 1) * B_ * H_;
        milstm_step<<<dim3(H_ / 16, B_ / 16 / 8), 256, 0, stream>>>(
            hin, hout, cbuf, WhT, WyT, x16, biasPack, alphaPk, out, t);
    }
}
